// DecoderLayer_8211977470069
// MI455X (gfx1250) — matrix operand order repaired and checked by reading, not compiled, not run on hardware
//
#include <hip/hip_runtime.h>
#include <hip/hip_bf16.h>

typedef __attribute__((ext_vector_type(16))) __bf16 v16bf;
typedef __attribute__((ext_vector_type(8)))  float  v8f;

constexpr int cB  = 2;
constexpr int cS  = 2048;
constexpr int cD  = 1024;
constexpr int cH  = 16;
constexpr int cDK = 64;     // cD / cH
constexpr int cFF = 4096;
constexpr int cBS = cB * cS;
constexpr float cEPS = 1e-6f;

union FragB16 { v16bf v; uint4 q[2]; };

// ---------------------------------------------------------------------------
// CDNA5 async copy: global -> LDS, 16 bytes per lane, tracked by ASYNCcnt.
// Generic shared pointers hold the raw LDS byte address in their low 32 bits
// (ISA 10.2: LDS aperture uses addr[31:0]); pass that as the VDST operand.
// ---------------------------------------------------------------------------
__device__ __forceinline__ void async_ld16(const void* lds, const void* g) {
  unsigned l = (unsigned)(size_t)lds;
  asm volatile("global_load_async_to_lds_b128 %0, %1, off"
               :: "v"(l), "v"(g) : "memory");
}
template <int N>
__device__ __forceinline__ void wait_async() {
  asm volatile("s_wait_asynccnt %0" :: "n"(N) : "memory");
}

// ---------------------------------------------------------------------------
// LayerNorm over last dim (D=1024), output bf16. One block (256 thr) per row.
// ---------------------------------------------------------------------------
__global__ __launch_bounds__(256)
void ln_bf16_kernel(const float* __restrict__ x,
                    const float* __restrict__ g,
                    const float* __restrict__ b,
                    __hip_bfloat16* __restrict__ out) {
  __shared__ float red[256];
  const int row = blockIdx.x;
  const int t   = threadIdx.x;
  const float* xr = x + (size_t)row * cD;

  float v[4];
  float s = 0.f;
  #pragma unroll
  for (int i = 0; i < 4; ++i) { v[i] = xr[t + i * 256]; s += v[i]; }
  red[t] = s; __syncthreads();
  for (int k = 128; k > 0; k >>= 1) { if (t < k) red[t] += red[t + k]; __syncthreads(); }
  const float mu = red[0] * (1.0f / cD);
  __syncthreads();

  float s2 = 0.f;
  #pragma unroll
  for (int i = 0; i < 4; ++i) { float d = v[i] - mu; s2 += d * d; }
  red[t] = s2; __syncthreads();
  for (int k = 128; k > 0; k >>= 1) { if (t < k) red[t] += red[t + k]; __syncthreads(); }
  const float rstd = rsqrtf(red[0] * (1.0f / cD) + cEPS);

  __hip_bfloat16* orow = out + (size_t)row * cD;
  #pragma unroll
  for (int i = 0; i < 4; ++i) {
    const int c = t + i * 256;
    orow[c] = __float2bfloat16((v[i] - mu) * rstd * g[c] + b[c]);
  }
}

// ---------------------------------------------------------------------------
// Row softmax over S=2048 f32 scores -> bf16 probs. One block per row.
// ---------------------------------------------------------------------------
__global__ __launch_bounds__(256)
void softmax_bf16_kernel(const float* __restrict__ sc,
                         __hip_bfloat16* __restrict__ pr) {
  __shared__ float red[256];
  const int row = blockIdx.x;
  const int t   = threadIdx.x;
  const float* sr = sc + (size_t)row * cS;

  float v[8];
  float m = -3.4e38f;
  #pragma unroll
  for (int i = 0; i < 8; ++i) { v[i] = sr[t + i * 256]; m = fmaxf(m, v[i]); }
  red[t] = m; __syncthreads();
  for (int k = 128; k > 0; k >>= 1) { if (t < k) red[t] = fmaxf(red[t], red[t + k]); __syncthreads(); }
  const float rm = red[0];
  __syncthreads();

  float s = 0.f;
  #pragma unroll
  for (int i = 0; i < 8; ++i) { v[i] = __expf(v[i] - rm); s += v[i]; }
  red[t] = s; __syncthreads();
  for (int k = 128; k > 0; k >>= 1) { if (t < k) red[t] += red[t + k]; __syncthreads(); }
  const float inv = 1.0f / red[0];

  __hip_bfloat16* prow = pr + (size_t)row * cS;
  #pragma unroll
  for (int i = 0; i < 8; ++i) prow[t + i * 256] = __float2bfloat16(v[i] * inv);
}

// ---------------------------------------------------------------------------
// f32 [K,N] -> bf16 [N,K] transposed cast (weights), 32x32 LDS tile.
// ---------------------------------------------------------------------------
__global__ void transpose_cast_kernel(const float* __restrict__ W,
                                      __hip_bfloat16* __restrict__ Wt,
                                      int K, int N) {
  __shared__ float tile[32][33];
  const int tx = threadIdx.x, ty = threadIdx.y;           // 32 x 8
  const int n0 = blockIdx.x * 32, k0 = blockIdx.y * 32;
  #pragma unroll
  for (int i = 0; i < 32; i += 8)
    tile[ty + i][tx] = W[(size_t)(k0 + ty + i) * N + (n0 + tx)];
  __syncthreads();
  #pragma unroll
  for (int i = 0; i < 32; i += 8)
    Wt[(size_t)(n0 + ty + i) * K + (k0 + tx)] = __float2bfloat16(tile[tx][ty + i]);
}

// ---------------------------------------------------------------------------
// LDS-staged, double-buffered WMMA GEMM:
//   C[M,N] = alpha * A[M,K] * Bt[N,K]^T (+bias) (+relu) (+residual)
// A, Bt bf16 row-major. 256 thr = 8 waves (4 M x 2 N); wave tile 32 x (16*WN)
// via 2 x WN v_wmma_f32_16x16x32_bf16 accumulators. Block tile 128 x 32*WN.
// K-step 32; A/B tiles staged in LDS with global_load_async_to_lds_b128
// (double buffered, s_wait_asynccnt handshake). LDS rows padded to 80 B so
// both A-fragment and B-fragment (16 B) reads are bank-conflict-free.
// mode 0: out[m*ldc+n] (outf f32 and/or outh bf16)
// mode 1: split heads  -> outh[((b*H+h)*S+s)*DK+dk]   (m=b*S+s, n=h*DK+dk)
// mode 2: split+transpose -> outh[((b*H+h)*DK+dk)*S+s]  (for V^T)
// ---------------------------------------------------------------------------
template <int WN>
__global__ __launch_bounds__(256)
void wmma_gemm_lds(const __hip_bfloat16* __restrict__ A, int lda,
                   const __hip_bfloat16* __restrict__ Bt, int ldb,
                   const float* __restrict__ bias,
                   const float* __restrict__ res, int ldr,
                   float* __restrict__ outf,
                   __hip_bfloat16* __restrict__ outh, int ldc,
                   int K, float alpha, int relu, int mode) {
  constexpr int BM   = 128;
  constexpr int BN   = 32 * WN;
  constexpr int LSTR = 80;                       // padded pitch of a 32-elem bf16 row
  constexpr int ASZ  = BM * LSTR;
  constexpr int BSZ  = BN * LSTR;
  constexpr int AOPS = (BM * 4) / 256;           // 16B-chunk async ops per thread (A)
  constexpr int BOPS = (BN * 4) / 256;           // 16B-chunk async ops per thread (B)
  constexpr int PF_OPS = AOPS + BOPS;
  __shared__ __align__(16) unsigned char smem[2 * (ASZ + BSZ)];

  const int tid   = threadIdx.x;
  const int lane  = tid & 31;
  const int wave  = tid >> 5;
  const int wm    = wave & 3;                    // wave row 0..3
  const int wn    = wave >> 2;                   // wave col 0..1
  const int l16   = lane & 15;
  const int lh    = lane >> 4;                   // lane half (0/1)
  const int mtile = blockIdx.y * BM;
  const int ntile = blockIdx.x * BN;

  auto prefetch = [&](int buf, int k0) {
    unsigned char* base = smem + buf * (ASZ + BSZ);
    #pragma unroll
    for (int q = 0; q < AOPS; ++q) {
      const int idx = tid + q * 256;
      const int row = idx >> 2, c = idx & 3;     // 4 x 16B chunks per 32-elem row
      async_ld16(base + row * LSTR + c * 16,
                 A + (size_t)(mtile + row) * lda + k0 + c * 8);
    }
    #pragma unroll
    for (int q = 0; q < BOPS; ++q) {
      const int idx = tid + q * 256;
      const int row = idx >> 2, c = idx & 3;
      async_ld16(base + ASZ + row * LSTR + c * 16,
                 Bt + (size_t)(ntile + row) * ldb + k0 + c * 8);
    }
  };

  v8f acc[2][WN] = {};
  prefetch(0, 0);
  int buf = 0;
  for (int k0 = 0; k0 < K; k0 += 32) {
    if (k0 + 32 < K) {
      prefetch(buf ^ 1, k0 + 32);   // fill other buffer
      wait_async<PF_OPS>();         // in-order completion => current tile resident
    } else {
      wait_async<0>();
    }
    __syncthreads();                // every wave's async writes now visible

    const unsigned char* base = smem + buf * (ASZ + BSZ);
    FragB16 a[2], b[WN];
    #pragma unroll
    for (int i = 0; i < 2; ++i) {
      const unsigned char* p = base + (wm * 32 + i * 16 + l16) * LSTR + lh * 16;
      a[i].q[0] = *(const uint4*)(p);        // K = lh*8 .. lh*8+7
      a[i].q[1] = *(const uint4*)(p + 32);   // K = 16+lh*8 .. 16+lh*8+7
    }
    #pragma unroll
    for (int j = 0; j < WN; ++j) {
      const unsigned char* p = base + ASZ + (wn * (WN * 16) + j * 16 + l16) * LSTR + lh * 16;
      b[j].q[0] = *(const uint4*)(p);        // K = lh*8 .. lh*8+7
      b[j].q[1] = *(const uint4*)(p + 32);   // K = 16+lh*8 .. 16+lh*8+7
    }
    #pragma unroll
    for (int i = 0; i < 2; ++i)
      #pragma unroll
      for (int j = 0; j < WN; ++j)
        acc[i][j] = __builtin_amdgcn_wmma_f32_16x16x32_bf16(
            false, a[i].v, false, b[j].v, (short)0, acc[i][j], false, false);
    __syncthreads();                // reads done before buffer is overwritten
    buf ^= 1;
  }

  // Epilogue. C/D layout: VGPR r -> M = r + 8*lh ; lane%16 -> N.
  #pragma unroll
  for (int i = 0; i < 2; ++i) {
    #pragma unroll
    for (int j = 0; j < WN; ++j) {
      #pragma unroll
      for (int r = 0; r < 8; ++r) {
        const int m = mtile + wm * 32 + i * 16 + r + lh * 8;
        const int n = ntile + wn * (WN * 16) + j * 16 + l16;
        float val = acc[i][j][r] * alpha;
        if (bias) val += bias[n];
        if (relu) val = fmaxf(val, 0.f);
        if (res)  val += res[(size_t)m * ldr + n];
        if (mode == 0) {
          if (outf) outf[(size_t)m * ldc + n] = val;
          if (outh) outh[(size_t)m * ldc + n] = __float2bfloat16(val);
        } else {
          const int bb = m >> 11;            // m / S  (S = 2048)
          const int s  = m & (cS - 1);
          const int h  = n >> 6;             // n / DK (DK = 64)
          const int dk = n & (cDK - 1);
          size_t idx;
          if (mode == 1) idx = (((size_t)(bb * cH + h)) * cS  + s ) * cDK + dk;
          else           idx = (((size_t)(bb * cH + h)) * cDK + dk) * cS  + s;
          outh[idx] = __float2bfloat16(val);
        }
      }
    }
  }
}

// ---------------------------------------------------------------------------
extern "C" void kernel_launch(void* const* d_in, const int* in_sizes, int n_in,
                              void* d_out, int out_size, void* d_ws, size_t ws_size,
                              hipStream_t stream) {
  const float* x     = (const float*)d_in[0];
  const float* Wq    = (const float*)d_in[1];
  const float* bq    = (const float*)d_in[2];
  const float* Wk    = (const float*)d_in[3];
  const float* bk    = (const float*)d_in[4];
  const float* Wv    = (const float*)d_in[5];
  const float* bv    = (const float*)d_in[6];
  const float* Wo    = (const float*)d_in[7];
  const float* bo    = (const float*)d_in[8];
  const float* W1    = (const float*)d_in[9];
  const float* b1    = (const float*)d_in[10];
  const float* W2    = (const float*)d_in[11];
  const float* b2    = (const float*)d_in[12];
  const float* ln1_g = (const float*)d_in[13];
  const float* ln1_b = (const float*)d_in[14];
  const float* ln2_g = (const float*)d_in[15];
  const float* ln2_b = (const float*)d_in[16];
  float* out = (float*)d_out;

  // Workspace carve-out (256B aligned slabs).
  char* ws = (char*)d_ws;
  size_t off = 0;
  auto alloc = [&](size_t bytes) -> char* {
    char* p = ws + off;
    off += (bytes + 255) & ~(size_t)255;
    return p;
  };
  __hip_bfloat16* ln1   = (__hip_bfloat16*)alloc((size_t)cBS * cD * 2);
  __hip_bfloat16* qb    = (__hip_bfloat16*)alloc((size_t)cBS * cD * 2);   // [B,H,S,DK]
  __hip_bfloat16* kb    = (__hip_bfloat16*)alloc((size_t)cBS * cD * 2);   // [B,H,S,DK]
  __hip_bfloat16* vtb   = (__hip_bfloat16*)alloc((size_t)cBS * cD * 2);   // [B,H,DK,S]
  __hip_bfloat16* attn  = (__hip_bfloat16*)alloc((size_t)cBS * cD * 2);   // [B,S,D]
  __hip_bfloat16* ln2   = (__hip_bfloat16*)alloc((size_t)cBS * cD * 2);
  __hip_bfloat16* hb    = (__hip_bfloat16*)alloc((size_t)cBS * cFF * 2);  // [B*S,FF]
  float*          sc    = (float*)         alloc((size_t)cS * cS * 4);    // per-head scores
  __hip_bfloat16* pb    = (__hip_bfloat16*)alloc((size_t)cS * cS * 2);    // per-head probs
  float*          x1    = (float*)         alloc((size_t)cBS * cD * 4);
  __hip_bfloat16* wqt   = (__hip_bfloat16*)alloc((size_t)cD * cD * 2);
  __hip_bfloat16* wkt   = (__hip_bfloat16*)alloc((size_t)cD * cD * 2);
  __hip_bfloat16* wvt   = (__hip_bfloat16*)alloc((size_t)cD * cD * 2);
  __hip_bfloat16* wot   = (__hip_bfloat16*)alloc((size_t)cD * cD * 2);
  __hip_bfloat16* w1t   = (__hip_bfloat16*)alloc((size_t)cD * cFF * 2);   // [FF,D]
  __hip_bfloat16* w2t   = (__hip_bfloat16*)alloc((size_t)cFF * cD * 2);   // [D,FF]

  const dim3 tb(32, 8);
  // Weights -> transposed bf16 (y = x @ W  ==  x @ Wt^T with Wt[N,K]).
  transpose_cast_kernel<<<dim3(cD / 32,  cD / 32),  tb, 0, stream>>>(Wq, wqt, cD, cD);
  transpose_cast_kernel<<<dim3(cD / 32,  cD / 32),  tb, 0, stream>>>(Wk, wkt, cD, cD);
  transpose_cast_kernel<<<dim3(cD / 32,  cD / 32),  tb, 0, stream>>>(Wv, wvt, cD, cD);
  transpose_cast_kernel<<<dim3(cD / 32,  cD / 32),  tb, 0, stream>>>(Wo, wot, cD, cD);
  transpose_cast_kernel<<<dim3(cFF / 32, cD / 32),  tb, 0, stream>>>(W1, w1t, cD, cFF);
  transpose_cast_kernel<<<dim3(cD / 32,  cFF / 32), tb, 0, stream>>>(W2, w2t, cFF, cD);

  // WN=4: block tile 128x128 (all large GEMMs). WN=2: block tile 128x64 (P@V).
  auto gemm4 = [&](const __hip_bfloat16* A, int lda, const __hip_bfloat16* Bt, int ldb,
                   const float* bias, const float* res, int ldr,
                   float* of, __hip_bfloat16* oh, int ldc,
                   int M, int N, int K, float alpha, int relu, int mode) {
    wmma_gemm_lds<4><<<dim3(N / 128, M / 128), dim3(256), 0, stream>>>(
        A, lda, Bt, ldb, bias, res, ldr, of, oh, ldc, K, alpha, relu, mode);
  };
  auto gemm2 = [&](const __hip_bfloat16* A, int lda, const __hip_bfloat16* Bt, int ldb,
                   const float* bias, const float* res, int ldr,
                   float* of, __hip_bfloat16* oh, int ldc,
                   int M, int N, int K, float alpha, int relu, int mode) {
    wmma_gemm_lds<2><<<dim3(N / 64, M / 128), dim3(256), 0, stream>>>(
        A, lda, Bt, ldb, bias, res, ldr, of, oh, ldc, K, alpha, relu, mode);
  };

  // Sub-layer 1: LN -> QKV -> attention -> Wo + residual
  ln_bf16_kernel<<<cBS, 256, 0, stream>>>(x, ln1_g, ln1_b, ln1);

  gemm4(ln1, cD, wqt, cD, bq, nullptr, 0, nullptr, qb,  0, cBS, cD, cD, 1.0f, 0, 1);
  gemm4(ln1, cD, wkt, cD, bk, nullptr, 0, nullptr, kb,  0, cBS, cD, cD, 1.0f, 0, 1);
  gemm4(ln1, cD, wvt, cD, bv, nullptr, 0, nullptr, vtb, 0, cBS, cD, cD, 1.0f, 0, 2);

  const float scale = 0.125f;  // 1/sqrt(DK=64)
  for (int bh = 0; bh < cB * cH; ++bh) {
    const __hip_bfloat16* qh  = qb  + (size_t)bh * cS * cDK;
    const __hip_bfloat16* kh  = kb  + (size_t)bh * cS * cDK;
    const __hip_bfloat16* vth = vtb + (size_t)bh * cDK * cS;
    const int b_ = bh / cH, h_ = bh % cH;

    // scores = scale * Qh @ Kh^T      [S,S] f32
    gemm4(qh, cDK, kh, cDK, nullptr, nullptr, 0, sc, nullptr, cS,
          cS, cS, cDK, scale, 0, 0);
    // probs = softmax(scores)          [S,S] bf16
    softmax_bf16_kernel<<<cS, 256, 0, stream>>>(sc, pb);
    // O_h = P @ V_h  (V^T supplies the [N,K] operand) -> attn[b, :, h*DK:]
    gemm2(pb, cS, vth, cS, nullptr, nullptr, 0,
          nullptr, attn + (size_t)b_ * cS * cD + (size_t)h_ * cDK, cD,
          cS, cDK, cS, 1.0f, 0, 0);
  }

  // x1 = x + attn @ Wo^T + bo
  gemm4(attn, cD, wot, cD, bo, x, cD, x1, nullptr, cD, cBS, cD, cD, 1.0f, 0, 0);

  // Sub-layer 2: LN -> FFN -> residual
  ln_bf16_kernel<<<cBS, 256, 0, stream>>>(x1, ln2_g, ln2_b, ln2);

  // h = relu(ln2 @ W1^T + b1)   (bf16 for next GEMM)
  gemm4(ln2, cD, w1t, cD, b1, nullptr, 0, nullptr, hb, cFF,
        cBS, cFF, cD, 1.0f, 1, 0);
  // out = x1 + h @ W2^T + b2
  gemm4(hb, cFF, w2t, cFF, b2, x1, cD, out, nullptr, cD,
        cBS, cD, cFF, 1.0f, 0, 0);
}
